// CausalSelfAttention_32263794327679
// MI455X (gfx1250) — matrix operand order repaired and checked by reading, not compiled, not run on hardware
//
#include <hip/hip_runtime.h>
#include <hip/hip_bf16.h>

// ---------------- problem constants (match reference) ----------------
constexpr int BB      = 2;
constexpr int TT      = 2048;
constexpr int DIM     = 2048;
constexpr int NH      = 16;    // H
constexpr int NKV     = 4;     // HKV
constexpr int HD      = 128;   // HEAD_DIM
constexpr int KV_DIM  = NKV * HD;   // 512
constexpr int GROUP   = NH / NKV;   // 4

typedef __attribute__((ext_vector_type(16))) __bf16 v16bf;
typedef __attribute__((ext_vector_type(8)))  float  v8f;

union Frag16 {
    v16bf v;
    uint4 q[2];
    unsigned short u[16];
};

__device__ __forceinline__ unsigned short f2bf(float x) {
    unsigned int u = __float_as_uint(x);
    unsigned int r = u + 0x7FFFu + ((u >> 16) & 1u);   // round-to-nearest-even
    return (unsigned short)(r >> 16);
}

__device__ __forceinline__ v8f wmma_bf16(v16bf a, v16bf b, v8f c) {
    return __builtin_amdgcn_wmma_f32_16x16x32_bf16(false, a, false, b, (short)0, c,
                                                   false, false);
}

// CDNA5 async copy: global -> LDS, 16B per lane, tracked by ASYNCcnt.
__device__ __forceinline__ void async_copy_b128(unsigned int lds_byte_addr, const void* g) {
    asm volatile("global_load_async_to_lds_b128 %0, %1, off"
                 :: "v"(lds_byte_addr), "v"((unsigned long long)(uintptr_t)g)
                 : "memory");
}

// ---------------- f32 -> bf16 elementwise ----------------
__global__ __launch_bounds__(256) void cvt_bf16_kernel(
    const float* __restrict__ src, unsigned short* __restrict__ dst, int n)
{
    int i = blockIdx.x * 256 + threadIdx.x;
    if (i < n) dst[i] = f2bf(src[i]);
}

// ---------------- WMMA GEMM: C[M,N] = A[M,K] * B[N,K]^T ----------------
// A, B bf16 row-major; C f32 row-major.
// Block = 4 waves -> 128(M) x 64(N) tile; wave -> 32x64.
// B tile (64 rows x 32 k) staged in LDS via async-to-LDS, triple buffered,
// shared by all 4 waves.
constexpr int BSTR = 40;   // LDS row stride in u16 (80 B: 16B-aligned, bank-spread)

__global__ __launch_bounds__(128) void gemm_bf16_wmma(
    const unsigned short* __restrict__ A,
    const unsigned short* __restrict__ Bw,
    float* __restrict__ C,
    int M, int N, int K)
{
    __shared__ __align__(16) unsigned short btile[3][64 * BSTR];

    const int tid  = threadIdx.x;
    const int lane = tid & 31;
    const int wave = tid >> 5;
    const int half = lane >> 4;
    const int l15  = lane & 15;
    const int n0   = blockIdx.x * 64;
    const int m0   = (blockIdx.y * 4 + wave) * 32;

    // B staging map: thread -> (row, k-half); covers 64 rows x 32 k as 2 x b128
    const int srow = tid >> 1;
    const int skh  = (tid & 1) * 16;            // k element offset (u16)
    const unsigned int lds0 =
        (unsigned int)(uintptr_t)(&btile[0][0]) + (unsigned int)(srow * BSTR + skh) * 2u;
    const unsigned short* sgb = Bw + (size_t)(n0 + srow) * K + skh;

    const unsigned short* arow0 = A + (size_t)(m0 +      l15) * K;
    const unsigned short* arow1 = A + (size_t)(m0 + 16 + l15) * K;

    v8f acc[2][4];
    #pragma unroll
    for (int s = 0; s < 2; ++s)
        #pragma unroll
        for (int nt = 0; nt < 4; ++nt) acc[s][nt] = (v8f){};

    // prologue: stage k-chunk 0 into buffer 0
    async_copy_b128(lds0,      sgb);
    async_copy_b128(lds0 + 16, sgb + 8);

    const int nk = K / 32;
    for (int it = 0; it < nk; ++it) {
        const int kc = it * 32;

        if (it + 1 < nk) {
            const unsigned int lb = lds0 + (unsigned int)(((it + 1) % 3) * 64 * BSTR * 2);
            const unsigned short* g = sgb + (kc + 32);
            async_copy_b128(lb,      g);
            async_copy_b128(lb + 16, g + 8);
            // allow the 2 just-issued copies to remain in flight; previous tile done
            asm volatile("s_wait_asynccnt 0x2" ::: "memory");
        } else {
            asm volatile("s_wait_asynccnt 0x0" ::: "memory");
        }
        __syncthreads();

        __builtin_prefetch(arow0 + kc + 256, 0, 1);
        __builtin_prefetch(arow1 + kc + 256, 0, 1);

        Frag16 a0, a1;
        a0.q[0] = *(const uint4*)(arow0 + kc +      half * 8);
        a0.q[1] = *(const uint4*)(arow0 + kc + 16 + half * 8);
        a1.q[0] = *(const uint4*)(arow1 + kc +      half * 8);
        a1.q[1] = *(const uint4*)(arow1 + kc + 16 + half * 8);

        const unsigned short* bl = &btile[it % 3][0];
        #pragma unroll
        for (int nt = 0; nt < 4; ++nt) {
            const unsigned short* p = bl + (nt * 16 + l15) * BSTR + half * 8;
            Frag16 bf;
            bf.q[0] = *(const uint4*)(p);
            bf.q[1] = *(const uint4*)(p + 16);
            acc[0][nt] = wmma_bf16(a0.v, bf.v, acc[0][nt]);
            acc[1][nt] = wmma_bf16(a1.v, bf.v, acc[1][nt]);
        }
    }

    #pragma unroll
    for (int s = 0; s < 2; ++s) {
        #pragma unroll
        for (int r = 0; r < 8; ++r) {
            const int m = m0 + s * 16 + r + 8 * half;
            float* crow = C + (size_t)m * N + n0 + l15;
            crow[0]  = acc[s][0][r];
            crow[16] = acc[s][1][r];
            crow[32] = acc[s][2][r];
            crow[48] = acc[s][3][r];
        }
    }
}

// ---------------- fused RMSNorm + RoPE (+gain*scale) for Q/K ----------------
// src: [B*T, nh*128] f32 ; dst: [(b*nh+h)*T + t]*128 + d  bf16
__global__ __launch_bounds__(128) void normrope_kernel(
    const float* __restrict__ src, unsigned short* __restrict__ dst,
    const float* __restrict__ gain, float extra_scale, int nh)
{
    const int gid  = blockIdx.x * 4 + (threadIdx.x >> 5);   // row over B*T*nh
    const int lane = threadIdx.x & 31;
    const int h  = gid % nh;
    const int bt = gid / nh;
    const int t  = bt % TT;
    const int b  = bt / TT;

    const float* row = src + (size_t)bt * (nh * HD) + h * HD;
    float e0 = row[lane], e1 = row[lane + 32], e2 = row[lane + 64], e3 = row[lane + 96];

    float ss = e0 * e0 + e1 * e1 + e2 * e2 + e3 * e3;
    #pragma unroll
    for (int i = 1; i < 32; i <<= 1) ss += __shfl_xor(ss, i, 32);

    float sc = rsqrtf(ss * (1.0f / 128.0f) + 1.1920929e-7f) * extra_scale;
    if (gain) sc *= gain[h];

    // inv_freq[i] = 10000^(-i/64), pair indices i0 = lane, i1 = lane+32
    const float kLn = 0.14391156831212787f;  // ln(10000)/64
    float a0 = (float)t * __expf(-(float)lane * kLn);
    float a1 = (float)t * __expf(-(float)(lane + 32) * kLn);
    float s0, c0, s1, c1;
    __sincosf(a0, &s0, &c0);
    __sincosf(a1, &s1, &c1);

    float o0 = ( e0 * c0 + e2 * s0) * sc;
    float o2 = (-e0 * s0 + e2 * c0) * sc;
    float o1 = ( e1 * c1 + e3 * s1) * sc;
    float o3 = (-e1 * s1 + e3 * c1) * sc;

    unsigned short* orow = dst + (((size_t)(b * nh + h)) * TT + t) * HD;
    orow[lane]      = f2bf(o0);
    orow[lane + 32] = f2bf(o1);
    orow[lane + 64] = f2bf(o2);
    orow[lane + 96] = f2bf(o3);
}

// ---------------- V: f32 [B*T, NKV*128] -> bf16 d-major [B,NKV,128,T] ----------------
__global__ __launch_bounds__(256) void vtrans_kernel(
    const float* __restrict__ vraw, unsigned short* __restrict__ vb)
{
    int idx = blockIdx.x * 256 + threadIdx.x;     // over B*T*KV_DIM
    int d  = idx % HD;
    int r  = idx / HD;
    int hk = r % NKV;
    int bt = r / NKV;
    int t  = bt % TT;
    int b  = bt / TT;
    vb[(((size_t)(b * NKV + hk)) * HD + d) * TT + t] = f2bf(vraw[idx]);
}

// ---------------- gate: sigmoid(x @ gate_w^T + gate_b) -> [B*T, H] f32 ----------------
__global__ __launch_bounds__(128) void gate_kernel(
    const float* __restrict__ x, const float* __restrict__ gw,
    const float* __restrict__ gb, float* __restrict__ gbuf)
{
    const int gid  = blockIdx.x * 4 + (threadIdx.x >> 5);  // over (B*T)*H
    const int lane = threadIdx.x & 31;
    const int h   = gid % NH;
    const int row = gid / NH;

    const float* xr = x  + (size_t)row * DIM;
    const float* wr = gw + (size_t)h * DIM;
    float acc = 0.f;
    for (int k = lane; k < DIM; k += 32) acc += xr[k] * wr[k];
    #pragma unroll
    for (int i = 1; i < 32; i <<= 1) acc += __shfl_xor(acc, i, 32);

    if (lane == 0) gbuf[gid] = 1.0f / (1.0f + __expf(-(acc + gb[h])));
}

// ---------------- causal GQA flash attention (bf16 WMMA, wave32) ----------------
// qb: [B,H,T,128] (scale & gain pre-folded)   kb: [B,HKV,T,128]
// vb: [B,HKV,128,T]   gbuf: [B*T,H] sigmoid   ybuf: [B*T, H*128] bf16
__global__ __launch_bounds__(128) void attn_kernel(
    const unsigned short* __restrict__ qb,
    const unsigned short* __restrict__ kb,
    const unsigned short* __restrict__ vb,
    const float* __restrict__ gbuf,
    unsigned short* __restrict__ ybuf)
{
    __shared__ __align__(16) unsigned short pstage[4][16 * 32];

    const int lane = threadIdx.x & 31;
    const int wave = threadIdx.x >> 5;
    const int half = lane >> 4;
    const int l15  = lane & 15;
    const int b  = blockIdx.z;
    const int h  = blockIdx.y;
    const int hk = h / GROUP;
    const int q0 = (blockIdx.x * 4 + wave) * 16;

    // Q fragments: 16x128 tile -> four 16x32 A-fragments
    const unsigned short* qrow = qb + (((size_t)(b * NH + h)) * TT + q0 + l15) * HD;
    Frag16 qf[4];
    #pragma unroll
    for (int c = 0; c < 4; ++c) {
        qf[c].q[0] = *(const uint4*)(qrow + c * 32 +      half * 8);
        qf[c].q[1] = *(const uint4*)(qrow + c * 32 + 16 + half * 8);
    }

    v8f o[8];
    #pragma unroll
    for (int d = 0; d < 8; ++d) o[d] = (v8f){};
    float mrow[8], lrow[8];
    #pragma unroll
    for (int r = 0; r < 8; ++r) { mrow[r] = -__builtin_inff(); lrow[r] = 0.f; }

    const unsigned short* kbase = kb + ((size_t)(b * NKV + hk)) * TT * HD;
    const unsigned short* vbase = vb + ((size_t)(b * NKV + hk)) * HD * TT;
    unsigned short* pb = pstage[wave];

    const int jmax = (q0 + 15) >> 5;
    for (int j = 0; j <= jmax; ++j) {
        const int kv0 = j * 32;

        // S = Q * K^T for this 16(q) x 32(kv) tile
        v8f s0 = (v8f){}, s1 = (v8f){};
        #pragma unroll
        for (int c = 0; c < 4; ++c) {
            const unsigned short* kr0 = kbase + (size_t)(kv0 +      l15) * HD + c * 32 + half * 8;
            const unsigned short* kr1 = kbase + (size_t)(kv0 + 16 + l15) * HD + c * 32 + half * 8;
            Frag16 kf0, kf1;
            kf0.q[0] = *(const uint4*)(kr0); kf0.q[1] = *(const uint4*)(kr0 + 16);
            kf1.q[0] = *(const uint4*)(kr1); kf1.q[1] = *(const uint4*)(kr1 + 16);
            s0 = wmma_bf16(qf[c].v, kf0.v, s0);
            s1 = wmma_bf16(qf[c].v, kf1.v, s1);
        }

        // online softmax (per-row state lives in one 16-lane half)
        #pragma unroll
        for (int r = 0; r < 8; ++r) {
            const int qg  = q0 + r + 8 * half;
            const int kvA = kv0 + l15;
            const int kvB = kv0 + 16 + l15;
            float sa = (kvA <= qg) ? s0[r] : -__builtin_inff();
            float sb = (kvB <= qg) ? s1[r] : -__builtin_inff();

            float mx = fmaxf(sa, sb);
            #pragma unroll
            for (int i = 1; i < 16; i <<= 1) mx = fmaxf(mx, __shfl_xor(mx, i, 32));

            float mn   = fmaxf(mrow[r], mx);
            float corr = __expf(mrow[r] - mn);
            float pa   = __expf(sa - mn);
            float pbv  = __expf(sb - mn);
            float rs   = pa + pbv;
            #pragma unroll
            for (int i = 1; i < 16; i <<= 1) rs += __shfl_xor(rs, i, 32);

            lrow[r] = lrow[r] * corr + rs;
            mrow[r] = mn;
            #pragma unroll
            for (int d = 0; d < 8; ++d) o[d][r] *= corr;

            const int m = r + 8 * half;
            pb[m * 32 +      l15] = f2bf(pa);
            pb[m * 32 + 16 + l15] = f2bf(pbv);
        }
        __threadfence_block();   // order LDS stores before cross-lane reload (same wave)

        // reload P as an A-fragment (C/D -> A layout transpose via LDS)
        Frag16 pf;
        pf.q[0] = *(const uint4*)(pb + l15 * 32 +      half * 8);
        pf.q[1] = *(const uint4*)(pb + l15 * 32 + 16 + half * 8);

        // O += P(16x32) * V(32x128)
        #pragma unroll
        for (int d = 0; d < 8; ++d) {
            const unsigned short* vr = vbase + (size_t)(d * 16 + l15) * TT + kv0 + half * 8;
            Frag16 vf;
            vf.q[0] = *(const uint4*)(vr);
            vf.q[1] = *(const uint4*)(vr + 16);
            o[d] = wmma_bf16(pf.v, vf.v, o[d]);
        }
    }

    // epilogue: 1/l and sigmoid gate folded, write bf16 [B*T, H*128]
    #pragma unroll
    for (int r = 0; r < 8; ++r) {
        const int m  = r + 8 * half;
        const int tq = q0 + m;
        const float g  = gbuf[(size_t)(b * TT + tq) * NH + h];
        const float sc = g / lrow[r];
        unsigned short* yr = ybuf + ((size_t)(b * TT + tq) * NH + h) * HD + l15;
        #pragma unroll
        for (int d = 0; d < 8; ++d) yr[d * 16] = f2bf(o[d][r] * sc);
    }
}

// ---------------- host orchestration ----------------
extern "C" void kernel_launch(void* const* d_in, const int* in_sizes, int n_in,
                              void* d_out, int out_size, void* d_ws, size_t ws_size,
                              hipStream_t stream)
{
    const float* x      = (const float*)d_in[0];
    const float* Wq     = (const float*)d_in[1];
    const float* Wk     = (const float*)d_in[2];
    const float* Wv     = (const float*)d_in[3];
    const float* Wproj  = (const float*)d_in[4];
    const float* q_gain = (const float*)d_in[5];
    const float* gate_w = (const float*)d_in[6];
    const float* gate_b = (const float*)d_in[7];

    constexpr size_t N_X   = (size_t)BB * TT * DIM;      // 8M
    constexpr size_t N_WQ  = (size_t)DIM * DIM;          // 4M
    constexpr size_t N_WK  = (size_t)KV_DIM * DIM;       // 1M
    constexpr size_t N_K   = (size_t)BB * TT * KV_DIM;   // 2M

    char* ws = (char*)d_ws;
    size_t off = 0;
    auto carve = [&](size_t bytes) -> char* {
        char* p = ws + off;
        off = (off + bytes + 255) & ~(size_t)255;
        return p;
    };

    unsigned short* xb   = (unsigned short*)carve(N_X  * 2);
    unsigned short* wqb  = (unsigned short*)carve(N_WQ * 2);
    unsigned short* wkb  = (unsigned short*)carve(N_WK * 2);
    unsigned short* wvb  = (unsigned short*)carve(N_WK * 2);
    unsigned short* wpb  = (unsigned short*)carve(N_WQ * 2);
    unsigned short* qbuf = (unsigned short*)carve(N_X  * 2);   // [B,H,T,128]
    unsigned short* kbuf = (unsigned short*)carve(N_K  * 2);   // [B,HKV,T,128]
    unsigned short* vbuf = (unsigned short*)carve(N_K  * 2);   // [B,HKV,128,T]
    float*          gbuf = (float*)carve((size_t)BB * TT * NH * 4);
    float*          qraw = (float*)carve(N_X * 4);             // also reused as ybuf
    float*          kraw = (float*)carve(N_K * 4);
    float*          vraw = (float*)carve(N_K * 4);
    unsigned short* ybuf = (unsigned short*)qraw;              // reuse (attention overwrites)

    // 1) bf16 conversions
    cvt_bf16_kernel<<<(int)((N_X  + 255) / 256), 256, 0, stream>>>(x,     xb,  (int)N_X);
    cvt_bf16_kernel<<<(int)((N_WQ + 255) / 256), 256, 0, stream>>>(Wq,    wqb, (int)N_WQ);
    cvt_bf16_kernel<<<(int)((N_WK + 255) / 256), 256, 0, stream>>>(Wk,    wkb, (int)N_WK);
    cvt_bf16_kernel<<<(int)((N_WK + 255) / 256), 256, 0, stream>>>(Wv,    wvb, (int)N_WK);
    cvt_bf16_kernel<<<(int)((N_WQ + 255) / 256), 256, 0, stream>>>(Wproj, wpb, (int)N_WQ);

    const int Mrows = BB * TT;  // 4096
    // 2) QKV projections (WMMA, async-LDS staged B)
    gemm_bf16_wmma<<<dim3(DIM / 64,    Mrows / 128), 128, 0, stream>>>(xb, wqb, qraw, Mrows, DIM,    DIM);
    gemm_bf16_wmma<<<dim3(KV_DIM / 64, Mrows / 128), 128, 0, stream>>>(xb, wkb, kraw, Mrows, KV_DIM, DIM);
    gemm_bf16_wmma<<<dim3(KV_DIM / 64, Mrows / 128), 128, 0, stream>>>(xb, wvb, vraw, Mrows, KV_DIM, DIM);

    // 3) RMSNorm + RoPE (+gain * 1/sqrt(128) folded for Q)
    normrope_kernel<<<(Mrows * NH) / 4,  128, 0, stream>>>(qraw, qbuf, q_gain,
                                                           0.08838834764831843f, NH);
    normrope_kernel<<<(Mrows * NKV) / 4, 128, 0, stream>>>(kraw, kbuf, nullptr, 1.0f, NKV);

    // 4) V transpose to d-major bf16
    vtrans_kernel<<<(int)(N_K / 256), 256, 0, stream>>>(vraw, vbuf);

    // 5) sigmoid gate
    gate_kernel<<<(Mrows * NH) / 4, 128, 0, stream>>>(x, gate_w, gate_b, gbuf);

    // 6) flash attention (overwrites qraw region as ybuf)
    attn_kernel<<<dim3(TT / 64, NH, BB), 128, 0, stream>>>(qbuf, kbuf, vbuf, gbuf, ybuf);

    // 7) output projection -> f32 d_out
    gemm_bf16_wmma<<<dim3(DIM / 64, Mrows / 128), 128, 0, stream>>>(ybuf, wpb, (float*)d_out,
                                                                    Mrows, DIM, DIM);
}
